// HybridSpectralAge_30863634989579
// MI455X (gfx1250) — compile-verified, matrix operand order checked by reading, not run on hardware
//
#include <hip/hip_runtime.h>
#include <hip/hip_bf16.h>
#include <math.h>

#define NB    4
#define NCPGS 20000
#define NFREQ 10001
#define NE    320000
#define NH    4
#define NHD   16
#define NGH   64
#define NFD   64
#define NFH   128
#define NPH   128

typedef __attribute__((ext_vector_type(16))) _Float16 v16h;
typedef __attribute__((ext_vector_type(8)))  float    v8f;

__device__ __forceinline__ float gelu_exact(float v) {
  return 0.5f * v * (1.0f + erff(v * 0.70710678118654752440f));
}

// blockDim.x must be a power of two
__device__ __forceinline__ float block_reduce_sum(float v, float* sh) {
  int tid = threadIdx.x;
  sh[tid] = v; __syncthreads();
  for (int s = blockDim.x >> 1; s > 0; s >>= 1) {
    if (tid < s) sh[tid] += sh[tid + s];
    __syncthreads();
  }
  float r = sh[0]; __syncthreads();
  return r;
}

// ---------------- per-row mean of x ----------------
__global__ void k_mean(const float* __restrict__ x, float* __restrict__ mean) {
  __shared__ float sh[256];
  int b = blockIdx.x, tid = threadIdx.x;
  float s = 0.f;
  for (int n = tid; n < NCPGS; n += 256) s += x[b * NCPGS + n];
  s = block_reduce_sum(s, sh);
  if (tid == 0) mean[b] = s / (float)NCPGS;
}

// ---------------- direct DFT magnitude: mag = log1p(|rfft(x-mean)|) ----------------
__global__ void __launch_bounds__(256) k_dft(const float* __restrict__ x,
                                             const float* __restrict__ mean,
                                             float* __restrict__ mag) {
  __shared__ float sh[256];
  int bk = blockIdx.x;
  int b = bk / NFREQ, k = bk % NFREQ;
  int tid = threadIdx.x;
  float mu = mean[b];
  // exact integer phase recurrence: r = (k*n) mod N
  int r    = (int)(((long long)k * tid) % NCPGS);
  int step = (int)(((long long)k * 256) % NCPGS);
  const float w = 6.28318530717958647692f / (float)NCPGS;
  float re = 0.f, im = 0.f;
  for (int n = tid; n < NCPGS; n += 256) {
    float v = x[b * NCPGS + n] - mu;
    float sn, cs;
    __sincosf((float)r * w, &sn, &cs);
    re += v * cs;
    im -= v * sn;
    r += step; if (r >= NCPGS) r -= NCPGS;
  }
  re = block_reduce_sum(re, sh);
  im = block_reduce_sum(im, sh);
  if (tid == 0) mag[b * NFREQ + k] = log1pf(sqrtf(re * re + im * im));
}

// ---------------- freq_w = softmax(fft_freq_weights) * NFREQ ----------------
__global__ void k_freq_softmax(const float* __restrict__ w, float* __restrict__ out) {
  __shared__ float sh[256];
  int tid = threadIdx.x;
  float mx = -3.4e38f;
  for (int i = tid; i < NFREQ; i += 256) mx = fmaxf(mx, w[i]);
  sh[tid] = mx; __syncthreads();
  for (int s = 128; s > 0; s >>= 1) { if (tid < s) sh[tid] = fmaxf(sh[tid], sh[tid + s]); __syncthreads(); }
  mx = sh[0]; __syncthreads();
  float sum = 0.f;
  for (int i = tid; i < NFREQ; i += 256) sum += expf(w[i] - mx);
  sum = block_reduce_sum(sum, sh);
  float scale = (float)NFREQ / sum;
  for (int i = tid; i < NFREQ; i += 256) out[i] = expf(w[i] - mx) * scale;
}

// ---------------- norm2[b] = sum_k mag[b][k]^2 ----------------
__global__ void k_norm2(const float* __restrict__ mag, float* __restrict__ norm2) {
  __shared__ float sh[256];
  int b = blockIdx.x, tid = threadIdx.x;
  float s = 0.f;
  for (int k = tid; k < NFREQ; k += 256) { float v = mag[b * NFREQ + k]; s += v * v; }
  s = block_reduce_sum(s, sh);
  if (tid == 0) norm2[b] = s;
}

// ---------------- fvec_pre[b][n] = sum_k mag*freqw*fft_w1[n][k] + b1[n] ----------------
__global__ void __launch_bounds__(256) k_fft1(const float* __restrict__ mag,
                                              const float* __restrict__ freqw,
                                              const float* __restrict__ w1,
                                              const float* __restrict__ b1,
                                              float* __restrict__ out) {
  __shared__ float sh[256];
  int bn = blockIdx.x, b = bn / NFH, n = bn % NFH;
  int tid = threadIdx.x;
  const float* mrow = mag + b * NFREQ;
  const float* wrow = w1 + (size_t)n * NFREQ;
  float s = 0.f;
  for (int k = tid; k < NFREQ; k += 256) s += mrow[k] * freqw[k] * wrow[k];
  s = block_reduce_sum(s, sh);
  if (tid == 0) out[b * NFH + n] = s + b1[n];
}

// ---------------- in-place LN(D=blockDim) then gelu ----------------
__global__ void k_ln_gelu(float* __restrict__ buf, const float* __restrict__ g,
                          const float* __restrict__ bb) {
  __shared__ float sh[128];
  int b = blockIdx.x, tid = threadIdx.x, D = blockDim.x;
  float v = buf[b * D + tid];
  float m = block_reduce_sum(v, sh) / (float)D;
  float d = v - m;
  float var = block_reduce_sum(d * d, sh) / (float)D;
  float y = d * rsqrtf(var + 1e-5f) * g[tid] + bb[tid];
  buf[b * D + tid] = gelu_exact(y);
}

// ---------------- out[b][n] = LN( (in*scale) @ W.T + bias ) , D = blockDim (64) ----------------
__global__ void k_mm_ln(const float* __restrict__ in, int K, float in_scale,
                        const float* __restrict__ W, const float* __restrict__ bias,
                        const float* __restrict__ g, const float* __restrict__ bb,
                        float* __restrict__ out) {
  __shared__ float sh[64];
  __shared__ float row[128];
  int b = blockIdx.x, tid = threadIdx.x, D = blockDim.x;
  for (int k = tid; k < K; k += D) row[k] = in[b * K + k] * in_scale;
  __syncthreads();
  float s = bias[tid];
  const float* wrow = W + (size_t)tid * K;
  for (int k = 0; k < K; ++k) s += row[k] * wrow[k];
  float m = block_reduce_sum(s, sh) / (float)D;
  float d = s - m;
  float var = block_reduce_sum(d * d, sh) / (float)D;
  out[b * D + tid] = d * rsqrtf(var + 1e-5f) * g[tid] + bb[tid];
}

// ---------------- plastic dots: y_slow and contrib pre-factor ----------------
__global__ void __launch_bounds__(256) k_pl(const float* __restrict__ mag,
                                            const float* __restrict__ norm2,
                                            const float* __restrict__ plW,
                                            const float* __restrict__ plA,
                                            const float* __restrict__ pl_bias,
                                            float* __restrict__ yslow,
                                            float* __restrict__ contrib) {
  __shared__ float sh[256];
  int bn = blockIdx.x, b = bn / NPH, n = bn % NPH;
  int tid = threadIdx.x;
  const float* mrow = mag + b * NFREQ;
  const float* wrow = plW + (size_t)n * NFREQ;
  const float* arow = plA + (size_t)n * NFREQ;
  float inv = 1.f / (sqrtf(norm2[b]) + 1e-8f);
  float s1 = 0.f, s2 = 0.f;
  for (int k = tid; k < NFREQ; k += 256) {
    float mv = mrow[k];
    s1 += mv * wrow[k];
    float sig = 1.f / (1.f + expf(-arow[k]));
    s2 += mv * mv * inv * sig;
  }
  s1 = block_reduce_sum(s1, sh);
  s2 = block_reduce_sum(s2, sh);
  if (tid == 0) { yslow[b * NPH + n] = s1 + pl_bias[n]; contrib[b * NPH + n] = s2; }
}

// ---------------- plastic = LN(gelu(y_slow + eta*tanh(y_slow)*contrib)) ----------------
__global__ void k_plastic(const float* __restrict__ yslow, const float* __restrict__ contrib,
                          const float* __restrict__ eta, const float* __restrict__ png,
                          const float* __restrict__ pnb, float* __restrict__ out) {
  __shared__ float sh[128];
  int b = blockIdx.x, tid = threadIdx.x;
  float ys = yslow[b * NPH + tid];
  float y = ys + eta[0] * tanhf(ys) * contrib[b * NPH + tid];
  float gv = gelu_exact(y);
  float m = block_reduce_sum(gv, sh) / (float)NPH;
  float d = gv - m;
  float var = block_reduce_sum(d * d, sh) / (float)NPH;
  out[b * NPH + tid] = d * rsqrtf(var + 1e-5f) * png[tid] + pnb[tid];
}

// ---------------- h0[b,n,:] = gelu(LN(x*ge_w + ge_b)) ----------------
__global__ void k_h0(const float* __restrict__ x, const float* __restrict__ gew,
                     const float* __restrict__ geb, const float* __restrict__ lng,
                     const float* __restrict__ lnb, float* __restrict__ h0) {
  __shared__ float sh[64];
  int id = blockIdx.x;              // b*NCPGS + node, x flat has same index
  int tid = threadIdx.x;            // channel 0..63
  float v = x[id] * gew[tid] + geb[tid];
  float m = block_reduce_sum(v, sh) / 64.f;
  float d = v - m;
  float var = block_reduce_sum(d * d, sh) / 64.f;
  float y = d * rsqrtf(var + 1e-5f) * lng[tid] + lnb[tid];
  h0[(size_t)id * 64 + tid] = gelu_exact(y);
}

// ---------------- h = h0 @ gat_w.T via v_wmma_f32_16x16x32_f16 ----------------
// M = NB*NCPGS = 80000, K = 64, N = 64. One 16x16 tile per wave, 8 waves/block.
__global__ void __launch_bounds__(256) k_gat_wmma(const float* __restrict__ h0,
                                                  const float* __restrict__ gatw,
                                                  float* __restrict__ h) {
  int wave = threadIdx.x >> 5;
  int lane = threadIdx.x & 31;
  int tile = blockIdx.x * 8 + wave;          // 20000 tiles exactly
  int mtile = tile >> 2, ntile = tile & 3;
  int half = lane >> 4;
  int l15  = lane & 15;
  int mrow = mtile * 16 + l15;
  int ncol = ntile * 16 + l15;
  v8f acc = {};
  for (int kk = 0; kk < 64; kk += 32) {
    v16h a, bm;
#pragma unroll
    for (int e = 0; e < 16; ++e) {
      // ISA 16-bit A layout: K = 16*(e/8) + 8*(lane/16) + (e%8)
      int k = kk + ((e >> 3) << 4) + (half << 3) + (e & 7);
      a[e]  = (_Float16)h0[(size_t)mrow * 64 + k];
      bm[e] = (_Float16)gatw[(size_t)ncol * 64 + k];   // B[k][n] = gat_w[n][k]
    }
    acc = __builtin_amdgcn_wmma_f32_16x16x32_f16(false, a, false, bm,
                                                 (short)0, acc, false, false);
  }
#pragma unroll
  for (int r = 0; r < 8; ++r) {
    int m = mtile * 16 + r + (half << 3);    // C/D layout: m = r + 8*(lane/16)
    h[(size_t)m * 64 + ncol] = acc[r];
  }
}

// ---------------- s_src/s_dst per (b,node,head) ----------------
__global__ void k_scores(const float* __restrict__ h, const float* __restrict__ attn,
                         float* __restrict__ ssrc, float* __restrict__ sdst) {
  int t = blockIdx.x * 256 + threadIdx.x;
  if (t >= NB * NCPGS * NH) return;
  int hh = t & 3;
  int bn = t >> 2;
  const float* hrow = h + (size_t)bn * 64 + hh * 16;
  const float* as = attn + hh * 32;
  float s1 = 0.f, s2 = 0.f;
#pragma unroll
  for (int d = 0; d < 16; ++d) { float hv = hrow[d]; s1 += hv * as[d]; s2 += hv * as[16 + d]; }
  ssrc[t] = s1; sdst[t] = s2;
}

// ---------------- edge pass 1: running max (initial value 0, per jnp .at[].max) ----------------
__global__ void k_edge_max(const int* __restrict__ ei, const float* __restrict__ ssrc,
                           const float* __restrict__ sdst, float* __restrict__ mmax) {
  int t = blockIdx.x * 256 + threadIdx.x;
  if (t >= NE * NH) return;
  int e = t >> 2, hh = t & 3;
  int s = ei[e], d = ei[NE + e];
  for (int b = 0; b < NB; ++b) {
    float sc = ssrc[(b * NCPGS + s) * NH + hh] + sdst[(b * NCPGS + d) * NH + hh];
    sc = sc >= 0.f ? sc : 0.2f * sc;
    if (sc > 0.f)  // positive floats order like ints; init is 0.0f
      atomicMax((int*)&mmax[(b * NCPGS + d) * NH + hh], __float_as_int(sc));
  }
}

// ---------------- edge pass 2: ssum += exp(score - max[dst]) ----------------
__global__ void k_edge_sum(const int* __restrict__ ei, const float* __restrict__ ssrc,
                           const float* __restrict__ sdst, const float* __restrict__ mmax,
                           float* __restrict__ ssum) {
  int t = blockIdx.x * 256 + threadIdx.x;
  if (t >= NE * NH) return;
  int e = t >> 2, hh = t & 3;
  int s = ei[e], d = ei[NE + e];
  for (int b = 0; b < NB; ++b) {
    int di = (b * NCPGS + d) * NH + hh;
    float sc = ssrc[(b * NCPGS + s) * NH + hh] + sdst[di];
    sc = sc >= 0.f ? sc : 0.2f * sc;
    atomicAdd(&ssum[di], expf(sc - mmax[di]));
  }
}

// ---------------- edge pass 3: gat_out[dst] += h[src] * attn ----------------
__global__ void k_edge_msg(const int* __restrict__ ei, const float* __restrict__ ssrc,
                           const float* __restrict__ sdst, const float* __restrict__ mmax,
                           const float* __restrict__ ssum, const float* __restrict__ h,
                           float* __restrict__ gout) {
  int t = blockIdx.x * 256 + threadIdx.x;
  if (t >= NE * NH) return;
  int e = t >> 2, hh = t & 3;
  int s = ei[e], d = ei[NE + e];
  for (int b = 0; b < NB; ++b) {
    int di = (b * NCPGS + d) * NH + hh;
    float sc = ssrc[(b * NCPGS + s) * NH + hh] + sdst[di];
    sc = sc >= 0.f ? sc : 0.2f * sc;
    float a = expf(sc - mmax[di]) / (ssum[di] + 1e-8f);
    const float* hrow = h + ((size_t)(b * NCPGS + s)) * 64 + hh * 16;
    float* orow = gout + ((size_t)(b * NCPGS + d)) * 64 + hh * 16;
#pragma unroll
    for (int k = 0; k < 16; ++k) atomicAdd(&orow[k], hrow[k] * a);
  }
}

// ---------------- pooled[b][c] = sum_n (h0 + gelu(gat_out)) ----------------
__global__ void k_pool(const float* __restrict__ h0, const float* __restrict__ gout,
                       float* __restrict__ pooled) {
  __shared__ float acc[64];
  int b = blockIdx.y;
  int node = blockIdx.x * 256 + threadIdx.x;
  if (threadIdx.x < 64) acc[threadIdx.x] = 0.f;
  __syncthreads();
  if (node < NCPGS) {
    size_t base = ((size_t)b * NCPGS + node) * 64;
    for (int c = 0; c < 64; ++c) {
      float v = h0[base + c] + gelu_exact(gout[base + c]);
      atomicAdd(&acc[c], v);
    }
  }
  __syncthreads();
  if (threadIdx.x < 64) atomicAdd(&pooled[b * 64 + threadIdx.x], acc[threadIdx.x]);
}

// ---------------- gate + fuse + head (tiny; one lane per batch) ----------------
__global__ void k_final(const float* __restrict__ fftf, const float* __restrict__ gf,
                        const float* __restrict__ pf,
                        const float* __restrict__ gw1, const float* __restrict__ gb1,
                        const float* __restrict__ gw2, const float* __restrict__ gb2,
                        const float* __restrict__ hw1, const float* __restrict__ hb1,
                        const float* __restrict__ hw2, const float* __restrict__ hb2,
                        float* __restrict__ out) {
  int b = threadIdx.x;
  if (b >= NB) return;
  const float* f0 = fftf + b * 64;
  const float* f1 = gf + b * 64;
  const float* f2 = pf + b * 64;
  float g1[6];
  for (int j = 0; j < 6; ++j) {
    float s = gb1[j];
    const float* wr = gw1 + j * 192;
    for (int i = 0; i < 64; ++i) s += f0[i] * wr[i] + f1[i] * wr[64 + i] + f2[i] * wr[128 + i];
    g1[j] = gelu_exact(s);
  }
  float g[3];
  for (int c = 0; c < 3; ++c) {
    float s = gb2[c];
    for (int j = 0; j < 6; ++j) s += g1[j] * gw2[c * 6 + j];
    g[c] = s;
  }
  float mx = fmaxf(g[0], fmaxf(g[1], g[2]));
  float e0 = expf(g[0] - mx), e1 = expf(g[1] - mx), e2 = expf(g[2] - mx);
  float inv = 1.f / (e0 + e1 + e2);
  float w0 = e0 * inv, w1 = e1 * inv, w2 = e2 * inv;
  float acc = hb2[0];
  for (int j = 0; j < 32; ++j) {
    float s = hb1[j];
    const float* wr = hw1 + j * 64;
    for (int i = 0; i < 64; ++i) s += (f0[i] * w0 + f1[i] * w1 + f2[i] * w2) * wr[i];
    acc += gelu_exact(s) * hw2[j];
  }
  out[b] = acc;
}

extern "C" void kernel_launch(void* const* d_in, const int* in_sizes, int n_in,
                              void* d_out, int out_size, void* d_ws, size_t ws_size,
                              hipStream_t stream) {
  const float* x        = (const float*)d_in[0];
  const int*   ei       = (const int*)  d_in[1];
  const float* freq_w_p = (const float*)d_in[2];
  const float* fft_w1   = (const float*)d_in[3];
  const float* fft_b1   = (const float*)d_in[4];
  const float* fft_ln1g = (const float*)d_in[5];
  const float* fft_ln1b = (const float*)d_in[6];
  const float* fft_w2   = (const float*)d_in[7];
  const float* fft_b2   = (const float*)d_in[8];
  const float* fft_ln2g = (const float*)d_in[9];
  const float* fft_ln2b = (const float*)d_in[10];
  const float* ge_w     = (const float*)d_in[11];
  const float* ge_b     = (const float*)d_in[12];
  const float* ge_lng   = (const float*)d_in[13];
  const float* ge_lnb   = (const float*)d_in[14];
  const float* gat_w    = (const float*)d_in[15];
  const float* gat_attn = (const float*)d_in[16];
  const float* gp_w     = (const float*)d_in[17];
  const float* gp_b     = (const float*)d_in[18];
  const float* gp_lng   = (const float*)d_in[19];
  const float* gp_lnb   = (const float*)d_in[20];
  const float* pl_W     = (const float*)d_in[21];
  const float* pl_alpha = (const float*)d_in[22];
  const float* pl_eta   = (const float*)d_in[23];
  const float* pl_bias  = (const float*)d_in[24];
  const float* pn_g     = (const float*)d_in[25];
  const float* pn_b     = (const float*)d_in[26];
  const float* pp_w     = (const float*)d_in[27];
  const float* pp_b     = (const float*)d_in[28];
  const float* pp_lng   = (const float*)d_in[29];
  const float* pp_lnb   = (const float*)d_in[30];
  const float* gate_w1  = (const float*)d_in[31];
  const float* gate_b1  = (const float*)d_in[32];
  const float* gate_w2  = (const float*)d_in[33];
  const float* gate_b2  = (const float*)d_in[34];
  const float* head_w1  = (const float*)d_in[35];
  const float* head_b1  = (const float*)d_in[36];
  const float* head_w2  = (const float*)d_in[37];
  const float* head_b2  = (const float*)d_in[38];
  float* out = (float*)d_out;
  float* ws  = (float*)d_ws;

  // workspace layout (float offsets, 256B aligned chunks)
  size_t off = 0;
  auto alloc = [&](size_t n) { size_t r = off; off += (n + 63) & ~(size_t)63; return r; };
  size_t MEAN   = alloc(NB);
  size_t NORM2  = alloc(NB);
  size_t FREQW  = alloc(NFREQ);
  size_t MAG    = alloc((size_t)NB * NFREQ);
  size_t FVEC   = alloc(NB * NFH);
  size_t FFTF   = alloc(NB * NFD);
  size_t YSLOW  = alloc(NB * NPH);
  size_t CONTR  = alloc(NB * NPH);
  size_t PLAST  = alloc(NB * NPH);
  size_t PLF    = alloc(NB * NFD);
  size_t GF     = alloc(NB * NFD);
  size_t H0     = alloc((size_t)NB * NCPGS * 64);
  size_t HM     = alloc((size_t)NB * NCPGS * 64);
  size_t SSRC   = alloc((size_t)NB * NCPGS * NH);
  size_t SDST   = alloc((size_t)NB * NCPGS * NH);
  size_t ZBEG   = off;                    // zero-init region below
  size_t MMAX   = alloc((size_t)NB * NCPGS * NH);
  size_t SSUM   = alloc((size_t)NB * NCPGS * NH);
  size_t GOUT   = alloc((size_t)NB * NCPGS * 64);
  size_t POOLED = alloc(NB * 64);
  size_t ZEND   = off;
  (void)ws_size; (void)in_sizes; (void)n_in; (void)out_size;

  hipMemsetAsync(ws + ZBEG, 0, (ZEND - ZBEG) * sizeof(float), stream);

  // spectral front-end
  k_mean<<<NB, 256, 0, stream>>>(x, ws + MEAN);
  k_dft<<<NB * NFREQ, 256, 0, stream>>>(x, ws + MEAN, ws + MAG);
  k_freq_softmax<<<1, 256, 0, stream>>>(freq_w_p, ws + FREQW);
  k_norm2<<<NB, 256, 0, stream>>>(ws + MAG, ws + NORM2);

  // fft branch
  k_fft1<<<NB * NFH, 256, 0, stream>>>(ws + MAG, ws + FREQW, fft_w1, fft_b1, ws + FVEC);
  k_ln_gelu<<<NB, NFH, 0, stream>>>(ws + FVEC, fft_ln1g, fft_ln1b);
  k_mm_ln<<<NB, NFD, 0, stream>>>(ws + FVEC, NFH, 1.0f, fft_w2, fft_b2, fft_ln2g, fft_ln2b, ws + FFTF);

  // plastic branch
  k_pl<<<NB * NPH, 256, 0, stream>>>(ws + MAG, ws + NORM2, pl_W, pl_alpha, pl_bias,
                                     ws + YSLOW, ws + CONTR);
  k_plastic<<<NB, NPH, 0, stream>>>(ws + YSLOW, ws + CONTR, pl_eta, pn_g, pn_b, ws + PLAST);
  k_mm_ln<<<NB, NFD, 0, stream>>>(ws + PLAST, NPH, 1.0f, pp_w, pp_b, pp_lng, pp_lnb, ws + PLF);

  // graph branch
  k_h0<<<NB * NCPGS, 64, 0, stream>>>(x, ge_w, ge_b, ge_lng, ge_lnb, ws + H0);
  k_gat_wmma<<<(NB * NCPGS / 16) * 4 / 8, 256, 0, stream>>>(ws + H0, gat_w, ws + HM);
  k_scores<<<(NB * NCPGS * NH + 255) / 256, 256, 0, stream>>>(ws + HM, gat_attn, ws + SSRC, ws + SDST);
  k_edge_max<<<(NE * NH + 255) / 256, 256, 0, stream>>>(ei, ws + SSRC, ws + SDST, ws + MMAX);
  k_edge_sum<<<(NE * NH + 255) / 256, 256, 0, stream>>>(ei, ws + SSRC, ws + SDST, ws + MMAX, ws + SSUM);
  k_edge_msg<<<(NE * NH + 255) / 256, 256, 0, stream>>>(ei, ws + SSRC, ws + SDST, ws + MMAX,
                                                        ws + SSUM, ws + HM, ws + GOUT);
  k_pool<<<dim3((NCPGS + 255) / 256, NB), 256, 0, stream>>>(ws + H0, ws + GOUT, ws + POOLED);
  k_mm_ln<<<NB, NFD, 0, stream>>>(ws + POOLED, NGH, 1.0f / (float)NCPGS, gp_w, gp_b,
                                  gp_lng, gp_lnb, ws + GF);

  // fuse + head
  k_final<<<1, 32, 0, stream>>>(ws + FFTF, ws + GF, ws + PLF,
                                gate_w1, gate_b1, gate_w2, gate_b2,
                                head_w1, head_b1, head_w2, head_b2, out);
}
